// MultiHeadSelfAttLayer_9285719294125
// MI455X (gfx1250) — matrix operand order repaired and checked by reading, not compiled, not run on hardware
//
#include <hip/hip_runtime.h>
#include <hip/hip_bf16.h>

// ---------------------------------------------------------------------------
// MI455X (gfx1250) implementation.
// B=32, S=512, E=512, H=8, P=64, FF=1536.
// Attention is the "linear" variant: scores[b,h,s] = x[b,s,:]·(wq[h]@Ksum)/8,
// so Q/K GEMMs are folded into tiny per-(b,h) vectors. The three big GEMMs
// (V-proj, FFN1, FFN2) run on v_wmma_f32_16x16x32_bf16 with a bf16 hi/lo
// split of the fp32 operands (hi*hi + hi*lo + lo*hi) for near-fp32 accuracy.
// GEMM tile: 128x64x32 per 256-thread block (8 wave32), aligned b128 LDS
// fragment loads, packed b64 LDS staging stores for A, and a register
// prefetch pipeline so next-tile global loads overlap the WMMA phase.
// ---------------------------------------------------------------------------

#define BB 32
#define SS 512
#define EE 512
#define HH 8
#define PP 64
#define FF 1536
#define MROWS (BB * SS)        // 16384

typedef __attribute__((ext_vector_type(16))) __bf16          v16bf;
typedef __attribute__((ext_vector_type(8)))  float           v8f;
typedef __attribute__((ext_vector_type(8)))  unsigned short  us8;
typedef __attribute__((ext_vector_type(4)))  float           f4;
typedef __attribute__((ext_vector_type(2)))  unsigned int    uu2;

union Frag16 { v16bf v; us8 h[2]; unsigned short u[16]; };

__device__ __forceinline__ void split_bf16(float f, unsigned short& hi, unsigned short& lo) {
    unsigned u  = __float_as_uint(f);
    unsigned rh = u + 0x7FFFu + ((u >> 16) & 1u);          // RNE to bf16
    unsigned short h = (unsigned short)(rh >> 16);
    float hf = __uint_as_float(((unsigned)h) << 16);
    float l  = f - hf;
    unsigned ul = __float_as_uint(l);
    unsigned rl = ul + 0x7FFFu + ((ul >> 16) & 1u);
    hi = h;
    lo = (unsigned short)(rl >> 16);
}

__device__ __forceinline__ float block_sum256(float v, float* red) {
    int tid = threadIdx.x;
    red[tid] = v; __syncthreads();
    #pragma unroll
    for (int off = 128; off > 0; off >>= 1) {
        if (tid < off) red[tid] += red[tid + off];
        __syncthreads();
    }
    float r = red[0]; __syncthreads();
    return r;
}

__device__ __forceinline__ float block_max256(float v, float* red) {
    int tid = threadIdx.x;
    red[tid] = v; __syncthreads();
    #pragma unroll
    for (int off = 128; off > 0; off >>= 1) {
        if (tid < off) red[tid] = fmaxf(red[tid], red[tid + off]);
        __syncthreads();
    }
    float r = red[0]; __syncthreads();
    return r;
}

// ---------------------------------------------------------------------------
// K0: xsum[b,e] = sum_s x[b,s,e]; also zero the per-head sqnorm accumulator.
// ---------------------------------------------------------------------------
__global__ __launch_bounds__(256) void xsum_kernel(const float* __restrict__ x,
                                                   float* __restrict__ xsum,
                                                   float* __restrict__ sqnorm) {
    int b = blockIdx.x, tid = threadIdx.x;
    if (b == 0 && tid < 16) sqnorm[tid] = 0.f;
    for (int e = tid; e < EE; e += 256) {
        float a = 0.f;
        const float* xb = x + (size_t)b * SS * EE + e;
        for (int s = 0; s < SS; ++s) a += xb[(size_t)s * EE];
        xsum[b * EE + e] = a;
    }
}

// ---------------------------------------------------------------------------
// K0b: repack wv [H,E,P] -> wvr [E, H*P] (row-major B operand for the V GEMM).
// ---------------------------------------------------------------------------
__global__ __launch_bounds__(256) void repack_wv_kernel(const float* __restrict__ wv,
                                                        float* __restrict__ wvr) {
    int idx = blockIdx.x * 256 + threadIdx.x;     // over H*E*P = 262144
    int h = idx >> 15, rem = idx & 32767;
    int e = rem >> 6, p = rem & 63;
    wvr[e * (HH * PP) + h * PP + p] = wv[idx];
}

// ---------------------------------------------------------------------------
// K1: wqk[b,h,e] = 0.125 * sum_p wq[h,e,p] * (sum_e' xsum[b,e'] * wk[h,e',p])
// ---------------------------------------------------------------------------
__global__ __launch_bounds__(256) void wqk_kernel(const float* __restrict__ xsum,
                                                  const float* __restrict__ wk,
                                                  const float* __restrict__ wq,
                                                  float* __restrict__ wqk) {
    int b = blockIdx.x >> 3, h = blockIdx.x & 7, tid = threadIdx.x;
    __shared__ float xs[EE];
    __shared__ float ks[PP];
    xs[tid]       = xsum[b * EE + tid];
    xs[tid + 256] = xsum[b * EE + tid + 256];
    __syncthreads();
    if (tid < PP) {
        const float* wkp = wk + (size_t)h * EE * PP + tid;
        float a = 0.f;
        for (int e = 0; e < EE; ++e) a += xs[e] * wkp[(size_t)e * PP];
        ks[tid] = a;
    }
    __syncthreads();
    for (int e = tid; e < EE; e += 256) {
        const float* wqp = wq + (size_t)h * EE * PP + (size_t)e * PP;
        float a = 0.f;
        #pragma unroll
        for (int p = 0; p < PP; ++p) a += wqp[p] * ks[p];
        wqk[(b * HH + h) * EE + e] = 0.125f * a;    // 1/sqrt(P) folded in
    }
}

// ---------------------------------------------------------------------------
// K2: scores + softmax over S per (b,h); writes att to workspace [B,H,S] and
//     the first output (att_w transposed to [B,S,H]).
// ---------------------------------------------------------------------------
__global__ __launch_bounds__(256) void attn_kernel(const float* __restrict__ x,
                                                   const float* __restrict__ wqk,
                                                   float* __restrict__ att_ws,
                                                   float* __restrict__ att_out) {
    int b = blockIdx.x >> 3, h = blockIdx.x & 7, tid = threadIdx.x;
    __shared__ float w[EE];
    __shared__ float sc[SS];
    __shared__ float red[256];
    w[tid]       = wqk[(b * HH + h) * EE + tid];
    w[tid + 256] = wqk[(b * HH + h) * EE + tid + 256];
    __syncthreads();
    for (int s = tid; s < SS; s += 256) {
        const float* xr = x + ((size_t)b * SS + s) * EE;
        float a = 0.f;
        #pragma unroll 8
        for (int e = 0; e < EE; ++e) a += xr[e] * w[e];
        sc[s] = a;
    }
    __syncthreads();
    float m = -3.4e38f;
    for (int s = tid; s < SS; s += 256) m = fmaxf(m, sc[s]);
    m = block_max256(m, red);
    float sum = 0.f;
    for (int s = tid; s < SS; s += 256) { float e_ = expf(sc[s] - m); sc[s] = e_; sum += e_; }
    sum = block_sum256(sum, red);
    float inv = 1.f / sum;
    for (int s = tid; s < SS; s += 256) {
        float a = sc[s] * inv;
        att_ws[((size_t)b * HH + h) * SS + s] = a;
        att_out[((size_t)b * SS + s) * HH + h] = a;
    }
}

// ---------------------------------------------------------------------------
// WMMA GEMM: C[M,N] = A[M,K] @ B[K,N] via bf16 hi/lo split, fp32 accumulate.
// Block = 256 threads (8 wave32), tile 128x64x32. Wave w owns rows
// [w*16, w*16+16) of the tile and all four 16x16 N-subtiles (12 WMMA / wave
// per K-step). Register-prefetch pipeline: next tile's global loads are
// issued right after the staging barrier, overlapping the WMMA phase.
//   MODE 0: V-projection epilogue: scale by att[b,h,s], store, accumulate
//           per-head sum of squares (ds atomics -> global atomic add f32).
//   MODE 1: + bias, leaky_relu(0.2)
//   MODE 2: + bias
// ---------------------------------------------------------------------------
template <int MODE>
__global__ __launch_bounds__(256) void gemm_bf16split_kernel(
    const float* __restrict__ A, const float* __restrict__ Bm,
    const float* __restrict__ bias, float* __restrict__ C,
    const float* __restrict__ att, float* __restrict__ sqnorm,
    int M, int N, int K) {
    __shared__ alignas(16) unsigned short Ah[128 * 32], Al[128 * 32];
    __shared__ alignas(16) unsigned short Bh[64 * 32],  Bl[64 * 32];   // [n][k]
    __shared__ float hsum[8];

    const int tid  = threadIdx.x;
    const int lane = tid & 31, wid = tid >> 5;
    const int m0 = blockIdx.x * 128, n0 = blockIdx.y * 64;
    const int hlf = lane >> 4, l16 = lane & 15;

    // A staging: thread covers 16 consecutive floats of one 32-wide row
    const int ar = tid >> 1, ac = (tid & 1) * 16;
    // B staging: thread covers 4 consecutive n at rows kB and kB+16
    const int kB = tid >> 4, nb = (tid & 15) * 4;

    const float* arow = A  + (size_t)(m0 + ar) * K + ac;
    const float* brow = Bm + n0 + nb;

    v8f acc[4] = {{}, {}, {}, {}};

    // ---- prefetch tile 0 into registers ----
    f4 pa[4], pb[2];
    #pragma unroll
    for (int c4 = 0; c4 < 4; ++c4) pa[c4] = *(const f4*)(arow + c4 * 4);
    pb[0] = *(const f4*)(brow + (size_t)kB * N);
    pb[1] = *(const f4*)(brow + (size_t)(kB + 16) * N);

    for (int kk = 0; kk < K; kk += 32) {
        // ---- split prefetched registers -> LDS ----
        #pragma unroll
        for (int c4 = 0; c4 < 4; ++c4) {
            unsigned short h0, l0, h1, l1, h2, l2, h3, l3;
            split_bf16(pa[c4].x, h0, l0); split_bf16(pa[c4].y, h1, l1);
            split_bf16(pa[c4].z, h2, l2); split_bf16(pa[c4].w, h3, l3);
            uu2 hv, lv;
            hv.x = (unsigned)h0 | ((unsigned)h1 << 16);
            hv.y = (unsigned)h2 | ((unsigned)h3 << 16);
            lv.x = (unsigned)l0 | ((unsigned)l1 << 16);
            lv.y = (unsigned)l2 | ((unsigned)l3 << 16);
            *(uu2*)(Ah + ar * 32 + ac + c4 * 4) = hv;
            *(uu2*)(Al + ar * 32 + ac + c4 * 4) = lv;
        }
        #pragma unroll
        for (int i = 0; i < 2; ++i) {
            int kr = i * 16 + kB;
            unsigned short h_, l_;
            split_bf16(pb[i].x, h_, l_); Bh[(nb + 0) * 32 + kr] = h_; Bl[(nb + 0) * 32 + kr] = l_;
            split_bf16(pb[i].y, h_, l_); Bh[(nb + 1) * 32 + kr] = h_; Bl[(nb + 1) * 32 + kr] = l_;
            split_bf16(pb[i].z, h_, l_); Bh[(nb + 2) * 32 + kr] = h_; Bl[(nb + 2) * 32 + kr] = l_;
            split_bf16(pb[i].w, h_, l_); Bh[(nb + 3) * 32 + kr] = h_; Bl[(nb + 3) * 32 + kr] = l_;
        }
        __syncthreads();

        // ---- issue next tile's global loads (overlap with WMMA below) ----
        if (kk + 32 < K) {
            const float* an = arow + kk + 32;
            #pragma unroll
            for (int c4 = 0; c4 < 4; ++c4) pa[c4] = *(const f4*)(an + c4 * 4);
            pb[0] = *(const f4*)(brow + (size_t)(kk + 32 + kB) * N);
            pb[1] = *(const f4*)(brow + (size_t)(kk + 48 + kB) * N);
        }

        // ---- A fragment (aligned 2x b128 per hi/lo) ----
        // lane<16 -> K {0..7,16..23}; lane>=16 -> K {8..15,24..31}
        const int row = wid * 16 + l16;
        Frag16 ah, al;
        ah.h[0] = *(const us8*)(Ah + row * 32 + hlf * 8);
        ah.h[1] = *(const us8*)(Ah + row * 32 + 16 + hlf * 8);
        al.h[0] = *(const us8*)(Al + row * 32 + hlf * 8);
        al.h[1] = *(const us8*)(Al + row * 32 + 16 + hlf * 8);

        #pragma unroll
        for (int t = 0; t < 4; ++t) {
            // B fragment: N = lane&15; lane<16 -> K {0..7,16..23}; lane>=16 -> K {8..15,24..31}
            const int col = t * 16 + l16;
            Frag16 bh, bl;
            bh.h[0] = *(const us8*)(Bh + col * 32 + hlf * 8);
            bh.h[1] = *(const us8*)(Bh + col * 32 + 16 + hlf * 8);
            bl.h[0] = *(const us8*)(Bl + col * 32 + hlf * 8);
            bl.h[1] = *(const us8*)(Bl + col * 32 + 16 + hlf * 8);
            v8f a_ = acc[t];
            a_ = __builtin_amdgcn_wmma_f32_16x16x32_bf16(false, ah.v, false, bh.v, (short)0, a_, false, false);
            a_ = __builtin_amdgcn_wmma_f32_16x16x32_bf16(false, ah.v, false, bl.v, (short)0, a_, false, false);
            a_ = __builtin_amdgcn_wmma_f32_16x16x32_bf16(false, al.v, false, bh.v, (short)0, a_, false, false);
            acc[t] = a_;
        }
        __syncthreads();
    }

    if (MODE == 0) {
        if (tid < 8) hsum[tid] = 0.f;
        __syncthreads();
    }
    float ss[4] = {0.f, 0.f, 0.f, 0.f};
    #pragma unroll
    for (int t = 0; t < 4; ++t) {
        int n = n0 + t * 16 + l16;
        #pragma unroll
        for (int r = 0; r < 8; ++r) {
            int m = m0 + wid * 16 + hlf * 8 + r;   // D layout: VGPR r -> M=r (+8 hi half)
            float val = acc[t][r];
            if (MODE == 0) {
                int bidx = m >> 9, s = m & 511, h = n >> 6;
                val *= att[((size_t)bidx * HH + h) * SS + s];
                ss[t] += val * val;
            } else {
                val += bias[n];
                if (MODE == 1) val = (val < 0.f) ? 0.2f * val : val;
            }
            C[(size_t)m * N + n] = val;
        }
    }
    if (MODE == 0) {
        #pragma unroll
        for (int t = 0; t < 4; ++t)
            atomicAdd(&hsum[(n0 + t * 16) >> 6], ss[t]);    // ds_add_f32
        __syncthreads();
        if (tid < 8) atomicAdd(&sqnorm[tid], hsum[tid]);    // global_atomic_add_f32
    }
}

// ---------------------------------------------------------------------------
// K4: merged (p-major, h-minor interleave) * per-head rsqrt-norm + x residual,
//     then LayerNorm(ln0) -> hbuf. One block per row (b,s).
// ---------------------------------------------------------------------------
__global__ __launch_bounds__(256) void merge_ln0_kernel(const float* __restrict__ ho,
                                                        const float* __restrict__ sqnorm,
                                                        const float* __restrict__ x,
                                                        const float* __restrict__ g,
                                                        const float* __restrict__ bb,
                                                        float* __restrict__ hbuf) {
    int r = blockIdx.x, tid = threadIdx.x;
    __shared__ float red[256];
    __shared__ float rs[HH];
    if (tid < HH) rs[tid] = rsqrtf(fmaxf(sqnorm[tid], 1e-12f));
    __syncthreads();
    int j0 = tid, j1 = tid + 256;
    // merged[b,s, p*H + h] = ho[b,s, h*P + p]
    float v0 = ho[(size_t)r * EE + (j0 & 7) * PP + (j0 >> 3)] * rs[j0 & 7] + x[(size_t)r * EE + j0];
    float v1 = ho[(size_t)r * EE + (j1 & 7) * PP + (j1 >> 3)] * rs[j1 & 7] + x[(size_t)r * EE + j1];
    float mean = block_sum256(v0 + v1, red) * (1.f / EE);
    float d0 = v0 - mean, d1 = v1 - mean;
    float var = block_sum256(d0 * d0 + d1 * d1, red) * (1.f / EE);
    float inv = rsqrtf(var + 1e-3f);
    hbuf[(size_t)r * EE + j0] = d0 * inv * g[j0] + bb[j0];
    hbuf[(size_t)r * EE + j1] = d1 * inv * g[j1] + bb[j1];
}

// ---------------------------------------------------------------------------
// K5: row-wise L2 normalize p1 (FF=1536) in place.
// ---------------------------------------------------------------------------
__global__ __launch_bounds__(256) void p1norm_kernel(float* __restrict__ p1) {
    int r = blockIdx.x, tid = threadIdx.x;
    __shared__ float red[256];
    float v[6]; float ss = 0.f;
    #pragma unroll
    for (int i = 0; i < 6; ++i) {
        v[i] = p1[(size_t)r * FF + tid + 256 * i];
        ss += v[i] * v[i];
    }
    ss = block_sum256(ss, red);
    float sc = rsqrtf(fmaxf(ss, 1e-12f));
    #pragma unroll
    for (int i = 0; i < 6; ++i) p1[(size_t)r * FF + tid + 256 * i] = v[i] * sc;
}

// ---------------------------------------------------------------------------
// K7: out = LayerNorm(p2 + h, ln1). One block per row.
// ---------------------------------------------------------------------------
__global__ __launch_bounds__(256) void final_ln_kernel(const float* __restrict__ p2,
                                                       const float* __restrict__ hbuf,
                                                       const float* __restrict__ g,
                                                       const float* __restrict__ bb,
                                                       float* __restrict__ out) {
    int r = blockIdx.x, tid = threadIdx.x;
    __shared__ float red[256];
    int j0 = tid, j1 = tid + 256;
    float v0 = p2[(size_t)r * EE + j0] + hbuf[(size_t)r * EE + j0];
    float v1 = p2[(size_t)r * EE + j1] + hbuf[(size_t)r * EE + j1];
    float mean = block_sum256(v0 + v1, red) * (1.f / EE);
    float d0 = v0 - mean, d1 = v1 - mean;
    float var = block_sum256(d0 * d0 + d1 * d1, red) * (1.f / EE);
    float inv = rsqrtf(var + 1e-3f);
    out[(size_t)r * EE + j0] = d0 * inv * g[j0] + bb[j0];
    out[(size_t)r * EE + j1] = d1 * inv * g[j1] + bb[j1];
}

// ---------------------------------------------------------------------------
// Launch
// ---------------------------------------------------------------------------
extern "C" void kernel_launch(void* const* d_in, const int* in_sizes, int n_in,
                              void* d_out, int out_size, void* d_ws, size_t ws_size,
                              hipStream_t stream) {
    (void)in_sizes; (void)n_in; (void)out_size; (void)ws_size;
    const float* x     = (const float*)d_in[0];
    const float* wq    = (const float*)d_in[1];
    const float* wk    = (const float*)d_in[2];
    const float* wv    = (const float*)d_in[3];
    const float* ln0_g = (const float*)d_in[4];
    const float* ln0_b = (const float*)d_in[5];
    const float* w1    = (const float*)d_in[6];
    const float* b1    = (const float*)d_in[7];
    const float* w2    = (const float*)d_in[8];
    const float* b2    = (const float*)d_in[9];
    const float* ln1_g = (const float*)d_in[10];
    const float* ln1_b = (const float*)d_in[11];

    float* out_att   = (float*)d_out;                     // [B,S,H]
    float* out_final = (float*)d_out + BB * SS * HH;      // [B,S,E]

    // workspace layout (floats)
    float* ws      = (float*)d_ws;
    float* w_sq    = ws;                                  // 64   (8 used)
    float* w_xsum  = ws + 64;                             // 16384
    float* w_wqk   = w_xsum + BB * EE;                    // 131072
    float* w_att   = w_wqk + BB * HH * EE;                // 131072
    float* w_h     = w_att + BB * HH * SS;                // 8388608
    float* w_hop2  = w_h + (size_t)MROWS * EE;            // 8388608 (ho, then p2)
    float* w_p1    = w_hop2 + (size_t)MROWS * EE;         // 25165824
    float* w_wvr   = w_p1 + (size_t)MROWS * FF;           // 262144
    // total ~162 MB

    xsum_kernel<<<BB, 256, 0, stream>>>(x, w_xsum, w_sq);
    repack_wv_kernel<<<(HH * EE * PP) / 256, 256, 0, stream>>>(wv, w_wvr);
    wqk_kernel<<<BB * HH, 256, 0, stream>>>(w_xsum, wk, wq, w_wqk);
    attn_kernel<<<BB * HH, 256, 0, stream>>>(x, w_wqk, w_att, out_att);

    // V projection + att scaling + per-head global sq-norm
    gemm_bf16split_kernel<0><<<dim3(MROWS / 128, EE / 64), 256, 0, stream>>>(
        x, w_wvr, nullptr, w_hop2, w_att, w_sq, MROWS, EE, EE);

    merge_ln0_kernel<<<MROWS, 256, 0, stream>>>(w_hop2, w_sq, x, ln0_g, ln0_b, w_h);

    // FFN1: leaky_relu(h @ w1 + b1)
    gemm_bf16split_kernel<1><<<dim3(MROWS / 128, FF / 64), 256, 0, stream>>>(
        w_h, w1, b1, w_p1, nullptr, nullptr, MROWS, FF, EE);

    p1norm_kernel<<<MROWS, 256, 0, stream>>>(w_p1);

    // FFN2: p1n @ w2 + b2  (reuses ho buffer for p2)
    gemm_bf16split_kernel<2><<<dim3(MROWS / 128, EE / 64), 256, 0, stream>>>(
        w_p1, w2, b2, w_hop2, nullptr, nullptr, MROWS, EE, FF);

    final_ln_kernel<<<MROWS, 256, 0, stream>>>(w_hop2, w_h, ln1_g, ln1_b, out_final);
}
